// POI_trans_80642305950301
// MI455X (gfx1250) — matrix operand order repaired and checked by reading, not compiled, not run on hardware
//
#include <hip/hip_runtime.h>
#include <hip/hip_bf16.h>

// ---------------------------------------------------------------------------
// Problem constants (match the reference)
// ---------------------------------------------------------------------------
static constexpr int B_   = 128;
static constexpr int T_   = 64;
static constexpr int L_   = 16384;
static constexpr int KT_TOTAL = L_ / 32;   // 512 K-chunks of 32 for 16x16x32 WMMA
static constexpr float FUSE_WEIGHT = 0.5f;

typedef __attribute__((ext_vector_type(16))) _Float16 v16h;
typedef __attribute__((ext_vector_type(8)))  float    v8f;

// ---------------------------------------------------------------------------
// Workspace layout (bytes):
//   0     : idx_last   int[128]
//   512   : idx_prev   int[128]
//   1024  : mask       float[128]
//   1536  : mn1        float[128]   (row min of rows_last)
//   2048  : mx1        float[128]   (row max of rows_last)
//   2560  : g          float[2]     (global min/max of adjust2)
//   4096  : Afrag      _Float16[128*16384]   (4 MiB, WMMA A-fragment layout)
//   4 MiB+4096 : prob2 float[128*16384]      (8 MiB, GEMM accumulator)
// Total ~12 MiB + 4 KiB.
// ---------------------------------------------------------------------------
static constexpr size_t WS_AFRAG = 4096;
static constexpr size_t WS_PROB2 = 4096 + (size_t)B_ * L_ * sizeof(_Float16);

// ---------------------------------------------------------------------------
// K1: per-sample index extraction + mask
// ---------------------------------------------------------------------------
__global__ void prep_idx_kernel(const int* __restrict__ traj,
                                const int* __restrict__ traj_len,
                                int* __restrict__ idx_last,
                                int* __restrict__ idx_prev,
                                float* __restrict__ maskv) {
    int b = threadIdx.x;
    if (b < B_) {
        int tl = traj_len[b];
        idx_last[b] = traj[((size_t)b * T_ + (tl - 1)) * 2 + 1] - 1;
        int p = tl - 2; if (p < 0) p = 0;
        idx_prev[b] = traj[((size_t)b * T_ + p) * 2 + 1] - 1;
        maskv[b] = (tl >= 2) ? 1.0f : 0.0f;
    }
}

// ---------------------------------------------------------------------------
// K2: gather rows_prev and pack into WMMA 16-bit A-fragment layout (f16).
// Packed 16-bit A 16x32 layout: lane l: M = l%16, half h = l/16;
//   element e -> K = 16h + e
// Storage: Afrag[((mt*KT_TOTAL + kt)*32 + lane)*16 + e]
// so each GEMM lane loads its 16 f16 (32 bytes) contiguously.
// ---------------------------------------------------------------------------
__global__ __launch_bounds__(256) void pack_A_kernel(const float* __restrict__ attMap,
                                                     const int* __restrict__ idx_prev,
                                                     _Float16* __restrict__ Afrag) {
    int b  = blockIdx.x;          // 0..127, one gathered row per block
    int mt = b >> 4;              // M tile (16 rows per tile)
    int mr = b & 15;              // row within tile
    const float* __restrict__ row = attMap + (size_t)idx_prev[b] * (size_t)L_;
    for (int task = threadIdx.x; task < 2 * KT_TOTAL; task += 256) {
        int kt = task >> 1;
        int h  = task & 1;
        int kbase = kt * 32;
        v16h frag;
#pragma unroll
        for (int j = 0; j < 8; ++j) {
            frag[j]     = (_Float16)row[kbase + 16 * h + j];
            frag[8 + j] = (_Float16)row[kbase + 16 * h + 8 + j];
        }
        v16h* dst = (v16h*)(Afrag + (((size_t)(mt * KT_TOTAL + kt)) * 32
                                     + (size_t)(mr + 16 * h)) * 16);
        *dst = frag;
    }
}

// ---------------------------------------------------------------------------
// K3: min/max reductions. Blocks 0..127: row min/max of attMap_e[idx_last[b]].
// Block 128: global min/max of adjust2.
// ---------------------------------------------------------------------------
__global__ __launch_bounds__(256) void minmax_kernel(const float* __restrict__ attMap,
                                                     const float* __restrict__ adjust2,
                                                     const int* __restrict__ idx_last,
                                                     float* __restrict__ mn1,
                                                     float* __restrict__ mx1,
                                                     float* __restrict__ g) {
    __shared__ float smn[8], smx[8];
    int bid = blockIdx.x;
    const float* __restrict__ src =
        (bid < B_) ? (attMap + (size_t)idx_last[bid] * (size_t)L_) : adjust2;
    float mn = INFINITY, mx = -INFINITY;
    for (int i = threadIdx.x; i < L_; i += 256) {
        float v = src[i];
        mn = fminf(mn, v);
        mx = fmaxf(mx, v);
    }
#pragma unroll
    for (int off = 16; off; off >>= 1) {
        mn = fminf(mn, __shfl_xor(mn, off, 32));
        mx = fmaxf(mx, __shfl_xor(mx, off, 32));
    }
    if ((threadIdx.x & 31) == 0) { smn[threadIdx.x >> 5] = mn; smx[threadIdx.x >> 5] = mx; }
    __syncthreads();
    if (threadIdx.x == 0) {
        mn = smn[0]; mx = smx[0];
#pragma unroll
        for (int w = 1; w < 8; ++w) { mn = fminf(mn, smn[w]); mx = fmaxf(mx, smx[w]); }
        if (bid < B_) { mn1[bid] = mn; mx1[bid] = mx; }
        else          { g[0] = mn; g[1] = mx; }
    }
}

// ---------------------------------------------------------------------------
// K4: GEMM  prob2[128,16384] += rows_prev(f16) @ attMap_e(f16-converted)
// Block = 256 threads (8 waves). Wave w owns 16 output columns
// (n0 = blockIdx.x*128 + 16w) across ALL 128 output rows -> acc = 8 x v8f.
// blockIdx.y in {0,1} splits K in half; exactly 2 fp32 atomicAdd partials per
// element => bitwise deterministic (fp32 add is commutative).
// B fragment layout (16-bit B 32x16): lane l: N = l%16, element e: K = e + 16*(l/16).
// ---------------------------------------------------------------------------
__global__ __launch_bounds__(256) void gemm_kernel(const float* __restrict__ Bm,
                                                   const _Float16* __restrict__ Afrag,
                                                   float* __restrict__ prob2) {
    const int wave = (int)(threadIdx.x >> 5);
    const int lane = (int)(threadIdx.x & 31);
    const int lhi  = lane >> 4;      // 0/1
    const int llo  = lane & 15;
    const int n0   = blockIdx.x * 128 + wave * 16;
    const int kt0  = blockIdx.y * (KT_TOTAL / 2);
    const int kt1  = kt0 + (KT_TOTAL / 2);

    v8f acc[8] = {};   // 64 VGPRs of fp32 accumulators

    for (int kt = kt0; kt < kt1; ++kt) {
        // --- B fragment: 16 coalesced 64B row-segments, f32 -> f16 in regs ---
        const float* __restrict__ bp =
            Bm + ((size_t)(kt * 32 + 16 * lhi)) * (size_t)L_ + (size_t)(n0 + llo);
        v16h bfrag;
#pragma unroll
        for (int e = 0; e < 16; ++e) bfrag[e] = (_Float16)bp[(size_t)e * L_];

        // --- 8 A fragments (L2-resident, pre-packed) + 8 WMMAs ---
        const _Float16* __restrict__ abase = Afrag + ((size_t)kt * 32 + (size_t)lane) * 16;
#pragma unroll
        for (int mt = 0; mt < 8; ++mt) {
            v16h afrag = *(const v16h*)(abase + (size_t)mt * (size_t)KT_TOTAL * 32 * 16);
            acc[mt] = __builtin_amdgcn_wmma_f32_16x16x32_f16(
                false, afrag, false, bfrag, (short)0, acc[mt], false, false);
        }
    }

    // C/D layout: VGPR r, lanes 0-15: M=r; lanes 16-31: M=r+8; N = lane%16.
#pragma unroll
    for (int mt = 0; mt < 8; ++mt) {
#pragma unroll
        for (int r = 0; r < 8; ++r) {
            int m = mt * 16 + r + 8 * lhi;
            atomicAdd(&prob2[(size_t)m * L_ + (size_t)(n0 + llo)], acc[mt][r]);
        }
    }
}

// ---------------------------------------------------------------------------
// K5: per-row min/max of prob2 (in-block reduce) + final fuse:
//   out = prob1_m + 0.5 * mask * (w * prob2_m)
// ---------------------------------------------------------------------------
__global__ __launch_bounds__(256) void finalize_kernel(const float* __restrict__ attMap,
                                                       const float* __restrict__ adjust2,
                                                       const float* __restrict__ prob2,
                                                       const int* __restrict__ idx_last,
                                                       const float* __restrict__ maskv,
                                                       const float* __restrict__ mn1v,
                                                       const float* __restrict__ mx1v,
                                                       const float* __restrict__ g,
                                                       float* __restrict__ out) {
    __shared__ float smn[8], smx[8];
    int b = blockIdx.x;
    const float* __restrict__ p2 = prob2 + (size_t)b * (size_t)L_;

    float mn = INFINITY, mx = -INFINITY;
    for (int n = threadIdx.x; n < L_; n += 256) {
        float v = p2[n];
        mn = fminf(mn, v);
        mx = fmaxf(mx, v);
    }
#pragma unroll
    for (int off = 16; off; off >>= 1) {
        mn = fminf(mn, __shfl_xor(mn, off, 32));
        mx = fmaxf(mx, __shfl_xor(mx, off, 32));
    }
    if ((threadIdx.x & 31) == 0) { smn[threadIdx.x >> 5] = mn; smx[threadIdx.x >> 5] = mx; }
    __syncthreads();
    mn = smn[0]; mx = smx[0];
#pragma unroll
    for (int w = 1; w < 8; ++w) { mn = fminf(mn, smn[w]); mx = fmaxf(mx, smx[w]); }

    const float* __restrict__ rl = attMap + (size_t)idx_last[b] * (size_t)L_;
    float a1 = mn1v[b];
    float s1 = 1.0f / (mx1v[b] - a1);
    float s2 = 1.0f / (mx - mn);
    float gm = g[0];
    float gs = 1.0f / (g[1] - gm);
    float fm = FUSE_WEIGHT * maskv[b];
    float* __restrict__ ob = out + (size_t)b * (size_t)L_;

    for (int n = threadIdx.x; n < L_; n += 256) {
        float p1m = (rl[n] - a1) * s1;
        float w   = (adjust2[n] - gm) * gs;
        float p2m = (p2[n] - mn) * s2;
        ob[n] = p1m + fm * (w * p2m);
    }
}

// ---------------------------------------------------------------------------
// Launch
// ---------------------------------------------------------------------------
extern "C" void kernel_launch(void* const* d_in, const int* in_sizes, int n_in,
                              void* d_out, int out_size, void* d_ws, size_t ws_size,
                              hipStream_t stream) {
    (void)in_sizes; (void)n_in; (void)out_size; (void)ws_size;
    // d_in order: Final_output (unused numerically), attMap_e, traj, traj_len, adjust2
    const float* attMap   = (const float*)d_in[1];
    const int*   traj     = (const int*)d_in[2];
    const int*   traj_len = (const int*)d_in[3];
    const float* adjust2  = (const float*)d_in[4];
    float* out = (float*)d_out;

    char* ws = (char*)d_ws;
    int*   idx_last = (int*)(ws + 0);
    int*   idx_prev = (int*)(ws + 512);
    float* maskv    = (float*)(ws + 1024);
    float* mn1      = (float*)(ws + 1536);
    float* mx1      = (float*)(ws + 2048);
    float* g        = (float*)(ws + 2560);
    _Float16* Afrag = (_Float16*)(ws + WS_AFRAG);
    float*    prob2 = (float*)(ws + WS_PROB2);

    // Zero the GEMM accumulator (graph-capture legal).
    hipMemsetAsync(prob2, 0, (size_t)B_ * L_ * sizeof(float), stream);

    prep_idx_kernel<<<1, 128, 0, stream>>>(traj, traj_len, idx_last, idx_prev, maskv);
    pack_A_kernel<<<B_, 256, 0, stream>>>(attMap, idx_prev, Afrag);
    minmax_kernel<<<B_ + 1, 256, 0, stream>>>(attMap, adjust2, idx_last, mn1, mx1, g);
    gemm_kernel<<<dim3(L_ / 128, 2), 256, 0, stream>>>(attMap, Afrag, prob2);
    finalize_kernel<<<B_, 256, 0, stream>>>(attMap, adjust2, prob2, idx_last, maskv,
                                            mn1, mx1, g, out);
}
